// SwinMambaBlock_38319698215050
// MI455X (gfx1250) — matrix operand order repaired and checked by reading, not compiled, not run on hardware
//
#include <hip/hip_runtime.h>
#include <hip/hip_bf16.h>

typedef __attribute__((ext_vector_type(16))) __bf16 v16bf;
typedef __attribute__((ext_vector_type(8)))  float  v8f;

#define B_N    8
#define C_N    192
#define HW_N   3136
#define L_N    3136
#define M_N    (B_N * L_N)      // 25088 tokens
#define DIN    384
#define DST    16
#define DTR    12
#define PS     48               // padded x_proj output stride (44 -> 48)

// ---------- helpers ----------
static __device__ __forceinline__ unsigned short f32_to_bf16(float f) {
    unsigned int u = __float_as_uint(f);
    unsigned int r = (u + 0x7FFFu + ((u >> 16) & 1u)) >> 16;
    return (unsigned short)r;
}

union FragU { unsigned int u[8]; v16bf v; };

// A fragment: row-major [M,K] bf16, 16x32 tile at (m0,k0).
// lane<16: M=lane, K pairs {0,1..6,7, 16,17..22,23}; lane>=16: +8 on K.
static __device__ __forceinline__ v16bf load_A_frag(const unsigned short* A, int lda,
                                                    int m0, int k0, int lane) {
    FragU r;
    int m = m0 + (lane & 15);
    int kb = (lane & 16) >> 1;                 // 0 or 8
    const unsigned short* p = A + (size_t)m * lda + k0 + kb;
#pragma unroll
    for (int i = 0; i < 4; ++i) r.u[i]     = *(const unsigned int*)(p + 2 * i);
#pragma unroll
    for (int i = 0; i < 4; ++i) r.u[4 + i] = *(const unsigned int*)(p + 16 + 2 * i);
    return r.v;
}

// B fragment: weights W row-major [N,K] bf16 (B[k][n] = W[n][k]).
// lane: N=lane&15, K pairs {0,1..6,7, 16,17..22,23}; lane>=16: +8 on K.
static __device__ __forceinline__ v16bf load_B_frag(const unsigned short* W, int ldw,
                                                    int n0, int k0, int lane) {
    FragU r;
    int n = n0 + (lane & 15);
    const unsigned int* p = (const unsigned int*)(W + (size_t)n * ldw + k0 + ((lane & 16) >> 1));
#pragma unroll
    for (int i = 0; i < 8; ++i) r.u[i] = p[(i & 3) + ((i >> 2) << 3)];
    return r.v;
}

static __device__ __forceinline__ v8f wmma_bf16(v16bf a, v16bf b, v8f c) {
    return __builtin_amdgcn_wmma_f32_16x16x32_bf16(false, a, false, b,
                                                   (short)0, c, false, false);
}

// ---------- 0: weight conversion (fp32 -> bf16, pad x_proj N 44->48) ----------
__global__ void cvt_weights_kernel(const float* __restrict__ in_w,   // 768x192
                                   const float* __restrict__ xp_w,   // 44x384
                                   const float* __restrict__ out_w,  // 192x384
                                   unsigned short* __restrict__ win_bf,
                                   unsigned short* __restrict__ wxp_bf,   // 48x384
                                   unsigned short* __restrict__ wout_bf) {
    int i = blockIdx.x * blockDim.x + threadIdx.x;
    const int S0 = 768 * 192, S1 = 48 * 384, S2 = 192 * 384;
    if (i < S0) { win_bf[i] = f32_to_bf16(in_w[i]); return; }
    i -= S0;
    if (i < S1) {
        int n = i / 384;
        wxp_bf[i] = (n < 44) ? f32_to_bf16(xp_w[i]) : (unsigned short)0;
        return;
    }
    i -= S1;
    if (i < S2) wout_bf[i] = f32_to_bf16(out_w[i]);
}

// ---------- 1: LayerNorm over C=192, one wave per token, NCHW gather ----------
__global__ void ln_kernel(const float* __restrict__ x, const float* __restrict__ g,
                          const float* __restrict__ bta, unsigned short* __restrict__ xn) {
    int pos  = blockIdx.x * 8 + (threadIdx.x >> 5);
    int lane = threadIdx.x & 31;
    if (pos >= M_N) return;
    int b = pos / HW_N, hw = pos % HW_N;
    const float* xp = x + (size_t)b * C_N * HW_N + hw;
    float vals[6], sum = 0.f, sq = 0.f;
#pragma unroll
    for (int i = 0; i < 6; ++i) {
        int c = lane + 32 * i;
        float v = xp[(size_t)c * HW_N];
        vals[i] = v; sum += v; sq += v * v;
    }
#pragma unroll
    for (int off = 16; off; off >>= 1) {
        sum += __shfl_xor(sum, off, 32);
        sq  += __shfl_xor(sq,  off, 32);
    }
    float mu = sum * (1.0f / C_N);
    float var = sq * (1.0f / C_N) - mu * mu;
    float rstd = rsqrtf(var + 1e-5f);
    unsigned short* o = xn + (size_t)pos * C_N;
#pragma unroll
    for (int i = 0; i < 6; ++i) {
        int c = lane + 32 * i;
        o[c] = f32_to_bf16((vals[i] - mu) * rstd * g[c] + bta[c]);
    }
}

// ---------- generic 1x1 WMMA GEMM (used for x_proj, N=48) ----------
__global__ void gemm_bf16_kernel(const unsigned short* __restrict__ A,
                                 const unsigned short* __restrict__ W,
                                 float* __restrict__ Cmat,
                                 int M, int N, int K, int lda, int ldw, int ldc) {
    int lane = threadIdx.x & 31;
    int tiles_n = N >> 4;
    int tile = blockIdx.x * (blockDim.x >> 5) + (threadIdx.x >> 5);
    if (tile >= (M >> 4) * tiles_n) return;
    int m0 = (tile / tiles_n) << 4;
    int n0 = (tile % tiles_n) << 4;
    v8f acc = {};
    for (int k0 = 0; k0 < K; k0 += 32) {
        v16bf a = load_A_frag(A, lda, m0, k0, lane);
        v16bf b = load_B_frag(W, ldw, n0, k0, lane);
        acc = wmma_bf16(a, b, acc);
    }
    int n = n0 + (lane & 15);
    int mb = m0 + ((lane >> 4) << 3);
#pragma unroll
    for (int r = 0; r < 8; ++r)
        Cmat[(size_t)(mb + r) * ldc + n] = acc[r];
}

// ---------- 2x2 register-blocked WMMA GEMM (32x32 per wave) ----------
__global__ void gemm_bf16_2x2_kernel(const unsigned short* __restrict__ A,
                                     const unsigned short* __restrict__ W,
                                     float* __restrict__ Cmat,
                                     int M, int N, int K, int lda, int ldw, int ldc) {
    int lane = threadIdx.x & 31;
    int tiles_n = N >> 5;
    int tile = blockIdx.x * (blockDim.x >> 5) + (threadIdx.x >> 5);
    if (tile >= (M >> 5) * tiles_n) return;
    int m0 = (tile / tiles_n) << 5;
    int n0 = (tile % tiles_n) << 5;
    v8f c00 = {}, c01 = {}, c10 = {}, c11 = {};
    for (int k0 = 0; k0 < K; k0 += 32) {
        v16bf a0 = load_A_frag(A, lda, m0,      k0, lane);
        v16bf a1 = load_A_frag(A, lda, m0 + 16, k0, lane);
        v16bf b0 = load_B_frag(W, ldw, n0,      k0, lane);
        v16bf b1 = load_B_frag(W, ldw, n0 + 16, k0, lane);
        c00 = wmma_bf16(a0, b0, c00);
        c01 = wmma_bf16(a0, b1, c01);
        c10 = wmma_bf16(a1, b0, c10);
        c11 = wmma_bf16(a1, b1, c11);
    }
    int n  = lane & 15;
    int mb = (lane >> 4) << 3;
#pragma unroll
    for (int r = 0; r < 8; ++r) {
        Cmat[(size_t)(m0      + mb + r) * ldc + n0      + n] = c00[r];
        Cmat[(size_t)(m0      + mb + r) * ldc + n0 + 16 + n] = c01[r];
        Cmat[(size_t)(m0 + 16 + mb + r) * ldc + n0      + n] = c10[r];
        Cmat[(size_t)(m0 + 16 + mb + r) * ldc + n0 + 16 + n] = c11[r];
    }
}

// ---------- 3: causal depthwise conv (k=4) + SiLU -> u (fp32 + bf16) ----------
__global__ void conv_silu_kernel(const float* __restrict__ xz, const float* __restrict__ cw,
                                 const float* __restrict__ cb, float* __restrict__ u,
                                 unsigned short* __restrict__ u_bf) {
    int i = blockIdx.x * blockDim.x + threadIdx.x;
    if (i >= M_N * DIN) return;
    int m = i / DIN, d = i - m * DIN;
    int b = m / L_N, l = m - b * L_N;
    float s = cb[d];
#pragma unroll
    for (int j = 0; j < 4; ++j) {
        int ll = l - 3 + j;
        if (ll >= 0) s += cw[d * 4 + j] * xz[(size_t)(b * L_N + ll) * 768 + d];
    }
    float uv = s / (1.f + __expf(-s));     // silu
    u[i] = uv;
    u_bf[i] = f32_to_bf16(uv);
}

// ---------- 5: dt = softplus(proj[:, :12] @ dt_w^T + dt_b)  (K=12, VALU) ----------
__global__ void dt_kernel(const float* __restrict__ proj, const float* __restrict__ dtw,
                          const float* __restrict__ dtb, float* __restrict__ dt) {
    int m = blockIdx.x;
    int d = threadIdx.x;
    __shared__ float pr[DTR];
    if (d < DTR) pr[d] = proj[(size_t)m * PS + d];
    __syncthreads();
    float s = dtb[d];
#pragma unroll
    for (int r = 0; r < DTR; ++r) s += pr[r] * dtw[d * DTR + r];
    dt[(size_t)m * DIN + d] = (s > 20.f) ? s : log1pf(__expf(s));
}

// ---------- 6: selective scan, h[16] in regs, B/C staged in LDS per 32 steps ----------
__global__ void scan_kernel(const float* __restrict__ dt, const float* __restrict__ u,
                            const float* __restrict__ proj, const float* __restrict__ A_log,
                            float* __restrict__ ys) {
    int b = blockIdx.x;
    int d = blockIdx.y * 128 + threadIdx.x;
    __shared__ float Bs[32][DST];
    __shared__ float Cs[32][DST];
    float Ad[DST], h[DST];
#pragma unroll
    for (int s = 0; s < DST; ++s) { Ad[s] = -__expf(A_log[d * DST + s]); h[s] = 0.f; }
    size_t base = (size_t)b * L_N;
    for (int t0 = 0; t0 < L_N; t0 += 32) {
        __syncthreads();
        for (int i = threadIdx.x; i < 32 * DST; i += 128) {
            int tt = i >> 4, s = i & 15;
            size_t row = (base + t0 + tt) * PS;
            Bs[tt][s] = proj[row + DTR + s];
            Cs[tt][s] = proj[row + DTR + DST + s];
        }
        __syncthreads();
        for (int tt = 0; tt < 32; ++tt) {
            size_t m = base + t0 + tt;
            float dtv = dt[m * DIN + d];
            float du  = dtv * u[m * DIN + d];
            float acc = 0.f;
#pragma unroll
            for (int s = 0; s < DST; ++s) {
                float dA = __expf(dtv * Ad[s]);
                h[s] = fmaf(dA, h[s], du * Bs[tt][s]);
                acc  = fmaf(h[s], Cs[tt][s], acc);
            }
            ys[m * DIN + d] = acc;
        }
    }
}

// ---------- 7: y = (ys + u*D) * silu(z) -> bf16 ----------
__global__ void gate_kernel(const float* __restrict__ ys, const float* __restrict__ u,
                            const float* __restrict__ Dv, const float* __restrict__ xz,
                            unsigned short* __restrict__ y_bf) {
    int i = blockIdx.x * blockDim.x + threadIdx.x;
    if (i >= M_N * DIN) return;
    int m = i / DIN, d = i - m * DIN;
    float yv = ys[i] + u[i] * Dv[d];
    float zv = xz[(size_t)m * 768 + DIN + d];
    float zs = zv / (1.f + __expf(-zv));
    y_bf[i] = f32_to_bf16(yv * zs);
}

// ---------- 8: out_proj 2x2 WMMA GEMM + residual + NHWC->NCHW store ----------
__global__ void out_gemm_kernel(const unsigned short* __restrict__ Ybf,
                                const unsigned short* __restrict__ Wout,
                                const float* __restrict__ x, float* __restrict__ out) {
    const int M = M_N, N = C_N, K = DIN;
    int lane = threadIdx.x & 31;
    int tiles_n = N >> 5;
    int tile = blockIdx.x * (blockDim.x >> 5) + (threadIdx.x >> 5);
    if (tile >= (M >> 5) * tiles_n) return;
    int m0 = (tile / tiles_n) << 5;
    int n0 = (tile % tiles_n) << 5;
    v8f c00 = {}, c01 = {}, c10 = {}, c11 = {};
    for (int k0 = 0; k0 < K; k0 += 32) {
        v16bf a0 = load_A_frag(Ybf, K, m0,      k0, lane);
        v16bf a1 = load_A_frag(Ybf, K, m0 + 16, k0, lane);
        v16bf b0 = load_B_frag(Wout, K, n0,      k0, lane);
        v16bf b1 = load_B_frag(Wout, K, n0 + 16, k0, lane);
        c00 = wmma_bf16(a0, b0, c00);
        c01 = wmma_bf16(a0, b1, c01);
        c10 = wmma_bf16(a1, b0, c10);
        c11 = wmma_bf16(a1, b1, c11);
    }
    int nl = lane & 15;
    int mb = (lane >> 4) << 3;
#pragma unroll
    for (int r = 0; r < 8; ++r) {
#pragma unroll
        for (int q = 0; q < 4; ++q) {
            int m = m0 + ((q >> 1) << 4) + mb + r;
            int n = n0 + ((q & 1) << 4) + nl;
            float v = (q == 0) ? c00[r] : (q == 1) ? c01[r] : (q == 2) ? c10[r] : c11[r];
            int bb = m / L_N, hw = m - bb * L_N;
            size_t o = ((size_t)bb * C_N + n) * HW_N + hw;
            out[o] = v + x[o];
        }
    }
}

extern "C" void kernel_launch(void* const* d_in, const int* in_sizes, int n_in,
                              void* d_out, int out_size, void* d_ws, size_t ws_size,
                              hipStream_t stream) {
    const float* x     = (const float*)d_in[0];
    const float* ln_g  = (const float*)d_in[1];
    const float* ln_b  = (const float*)d_in[2];
    const float* in_w  = (const float*)d_in[3];
    const float* cw    = (const float*)d_in[4];
    const float* cb    = (const float*)d_in[5];
    const float* xp_w  = (const float*)d_in[6];
    const float* dtw   = (const float*)d_in[7];
    const float* dtb   = (const float*)d_in[8];
    const float* A_log = (const float*)d_in[9];
    const float* Dv    = (const float*)d_in[10];
    const float* out_w = (const float*)d_in[11];
    float* out = (float*)d_out;

    char* ws = (char*)d_ws;
    size_t off = 0;
    auto take = [&](size_t bytes) { size_t o = off; off += (bytes + 255) & ~size_t(255); return o; };
    unsigned short* xn_bf  = (unsigned short*)(ws + take((size_t)M_N * C_N * 2));
    unsigned short* win_bf = (unsigned short*)(ws + take(768 * 192 * 2));
    unsigned short* wxp_bf = (unsigned short*)(ws + take(48 * 384 * 2));
    unsigned short* wout_bf= (unsigned short*)(ws + take(192 * 384 * 2));
    float* xz   = (float*)(ws + take((size_t)M_N * 768 * 4));
    float* u    = (float*)(ws + take((size_t)M_N * DIN * 4));
    unsigned short* u_bf = (unsigned short*)(ws + take((size_t)M_N * DIN * 2));
    float* proj = (float*)(ws + take((size_t)M_N * PS * 4));
    float* dt   = (float*)(ws + take((size_t)M_N * DIN * 4));
    float* ys   = (float*)(ws + take((size_t)M_N * DIN * 4));
    unsigned short* y_bf = (unsigned short*)(ws + take((size_t)M_N * DIN * 2));

    // 0: weights -> bf16
    {
        int tot = 768 * 192 + 48 * 384 + 192 * 384;
        cvt_weights_kernel<<<(tot + 255) / 256, 256, 0, stream>>>(in_w, xp_w, out_w,
                                                                  win_bf, wxp_bf, wout_bf);
    }
    // 1: LayerNorm
    ln_kernel<<<M_N / 8, 256, 0, stream>>>(x, ln_g, ln_b, xn_bf);
    // 2: in_proj (2x2 blocked): (M,192) @ (768,192)^T -> xz (M,768)
    {
        int tiles = (M_N / 32) * (768 / 32);
        gemm_bf16_2x2_kernel<<<(tiles + 3) / 4, 128, 0, stream>>>(xn_bf, win_bf, xz,
                                                                  M_N, 768, 192, 192, 192, 768);
    }
    // 3: depthwise conv + SiLU -> u
    conv_silu_kernel<<<(M_N * DIN + 255) / 256, 256, 0, stream>>>(xz, cw, cb, u, u_bf);
    // 4: x_proj (1x1): (M,384) @ (48,384)^T -> proj (M,48)
    {
        int tiles = (M_N / 16) * (PS / 16);
        gemm_bf16_kernel<<<(tiles + 3) / 4, 128, 0, stream>>>(u_bf, wxp_bf, proj,
                                                              M_N, PS, 384, 384, 384, PS);
    }
    // 5: dt
    dt_kernel<<<M_N, DIN, 0, stream>>>(proj, dtw, dtb, dt);
    // 6: selective scan
    scan_kernel<<<dim3(B_N, DIN / 128), 128, 0, stream>>>(dt, u, proj, A_log, ys);
    // 7: gating
    gate_kernel<<<(M_N * DIN + 255) / 256, 256, 0, stream>>>(ys, u, Dv, xz, y_bf);
    // 8: out_proj (2x2) + residual + transpose
    {
        int tiles = (M_N / 32) * (C_N / 32);
        out_gemm_kernel<<<(tiles + 3) / 4, 128, 0, stream>>>(y_bf, wout_bf, x, out);
    }
}
